// StressNN_21921513079369
// MI455X (gfx1250) — matrix operand order repaired and checked by reading, not compiled, not run on hardware
//
#include <hip/hip_runtime.h>

// ---------------------------------------------------------------------------
// MI455X / gfx1250 stress-MLP: bf16 WMMA (V_WMMA_F32_16X16X32_BF16), wave32.
// One wave owns a 16-particle x 128-feature activation tile held in LDS.
// Weights are pre-converted to transposed bf16 in d_ws (L2/WGP$-resident).
// GELU uses a branch-free erf (A&S 7.1.26, |err|<1.5e-7) so the VALU work
// co-executes with the WMMA pipe instead of generating exec-mask branches.
// f32->bf16 uses the native hardware convert (RNE) via a __bf16 cast.
// ---------------------------------------------------------------------------

typedef __attribute__((ext_vector_type(16))) __bf16       v16bf;
typedef __attribute__((ext_vector_type(8)))  float        v8f;
typedef __attribute__((ext_vector_type(4)))  unsigned int v4u;

#define DEVINL __device__ __forceinline__

// Native f32 -> bf16 convert (fptrunc, round-to-nearest-even on gfx1250).
DEVINL unsigned short f2bf(float f) {
    union { __bf16 h; unsigned short u; } v;
    v.h = (__bf16)f;
    return v.u;
}

union Frag { v16bf v; v4u q[2]; };

// A-matrix 16x32 bf16 fragment (ISA 7.12.2): lane L holds row M = L&15.
// V0-3: K = kc*32 + (L>>4)*8 + [0..7] ; V4-7: same + 16.
DEVINL v16bf load_a_frag(const unsigned short* xw, int lane, int kc) {
    const int row = lane & 15;
    const int bk  = kc * 32 + ((lane >> 4) << 3);
    const unsigned short* p = xw + row * 136 + bk;
    Frag f;
    f.q[0] = *(const v4u*)(p);
    f.q[1] = *(const v4u*)(p + 16);
    return f.v;
}

// B-matrix 32x16 bf16 fragment: lane L holds column N = n0+(L&15);
// K = kc*32 + (L>>4)*8 + [0..7] and same + 16 (matches A).  Weights are stored
// transposed (Wt[n][k]) so these are two contiguous 16-byte row slices.
DEVINL v16bf load_b_frag(const unsigned short* wt, int wstr, int n0, int lane, int kc) {
    const int row = n0 + (lane & 15);
    const int bk  = kc * 32 + ((lane >> 4) << 3);
    const unsigned short* p = wt + row * wstr + bk;
    Frag f;
    f.q[0] = *(const v4u*)(p);
    f.q[1] = *(const v4u*)(p + 16);
    return f.v;
}

// Branch-free exact-GELU: erf via Abramowitz-Stegun 7.1.26 (|err| <= 1.5e-7,
// far below the bf16 activation quantization).  No exec-mask divergence:
// ~6 fma + v_rcp_f32 + v_exp_f32 per element, dual-issues with WMMA.
DEVINL float gelu_exact(float x) {
    const float z  = x * 0.7071067811865475f;
    const float az = fabsf(z);
    const float t  = __builtin_amdgcn_rcpf(fmaf(0.3275911f, az, 1.0f));
    float poly = fmaf(t, 1.061405429f, -1.453152027f);
    poly = fmaf(t, poly, 1.421413741f);
    poly = fmaf(t, poly, -0.284496736f);
    poly = fmaf(t, poly, 0.254829592f);
    poly *= t;
    const float e  = 1.0f - poly * __expf(-az * az);
    const float er = copysignf(e, z);
    return 0.5f * x * (1.0f + er);
}

// One dense layer: X[16xK] @ W[KxN=128] + b, optional GELU, result back to xw.
template <int KC, bool GELU>
DEVINL void mlp_layer(unsigned short* xw, const unsigned short* wt, int wstr,
                      const float* __restrict__ bias, int lane) {
    v16bf a[KC];
#pragma unroll
    for (int kc = 0; kc < KC; ++kc) a[kc] = load_a_frag(xw, lane, kc);

    const int nl = lane & 15;
    const int mh = (lane >> 4) << 3;
#pragma unroll
    for (int nt = 0; nt < 8; ++nt) {
        float bn = bias[nt * 16 + nl];
        v8f c;
#pragma unroll
        for (int g = 0; g < 8; ++g) c[g] = bn;
#pragma unroll
        for (int kc = 0; kc < KC; ++kc) {
            v16bf b = load_b_frag(wt, wstr, nt * 16, lane, kc);
            c = __builtin_amdgcn_wmma_f32_16x16x32_bf16(
                    false, a[kc], false, b, (short)0, c, false, false);
        }
        const int nb = nt * 16 + nl;
#pragma unroll
        for (int g = 0; g < 8; ++g) {
            float y = c[g];
            if (GELU) y = gelu_exact(y);
            xw[(g + mh) * 136 + nb] = f2bf(y);     // C/D layout: (m = g+8*(L>>4), n)
        }
    }
}

// ---------------------------------------------------------------------------
// Weight pre-conversion: fp32 -> bf16, transposed + padded into d_ws.
//   region0 @ 0      : Wt1 [128][96]   (K 89..95 zero)
//   region1 @ 12288  : Wt2..Wt6 [128][128] x5
//   region2 @ 94208  : Wt7 [16][128]   (rows 9..15 zero)
// ---------------------------------------------------------------------------
__global__ __launch_bounds__(256) void wconv_kernel(
    const float* __restrict__ W1, const float* __restrict__ W2,
    const float* __restrict__ W3, const float* __restrict__ W4,
    const float* __restrict__ W5, const float* __restrict__ W6,
    const float* __restrict__ W7, unsigned short* __restrict__ wt)
{
    int t = blockIdx.x * 256 + threadIdx.x;
    if (t < 128 * 96) {
        int n = t / 96, k = t - n * 96;
        wt[t] = f2bf(k < 89 ? W1[k * 128 + n] : 0.0f);
        return;
    }
    t -= 128 * 96;
    if (t < 5 * 128 * 128) {
        int l = t / 16384, r = t - l * 16384;
        int n = r / 128, k = r - n * 128;
        const float* W = (l == 0) ? W2 : (l == 1) ? W3 : (l == 2) ? W4
                                       : (l == 3) ? W5 : W6;
        wt[12288 + t] = f2bf(W[k * 128 + n]);
        return;
    }
    t -= 5 * 128 * 128;
    if (t < 16 * 128) {
        int n = t / 128, k = t - n * 128;
        wt[94208 + t] = f2bf(n < 9 ? W7[k * 9 + n] : 0.0f);
    }
}

// ---------------------------------------------------------------------------
// Fused strain-features + 7-layer MLP + symmetrize.
// 256 threads = 8 waves; each wave handles 16 particles with zero barriers.
// ---------------------------------------------------------------------------
__global__ __launch_bounds__(256) void stress_mlp_kernel(
    const float* __restrict__ F, const float* __restrict__ Cin,
    const int* __restrict__ trajp, const float* __restrict__ latent,
    const float* __restrict__ b1, const float* __restrict__ b2,
    const float* __restrict__ b3, const float* __restrict__ b4,
    const float* __restrict__ b5, const float* __restrict__ b6,
    const float* __restrict__ b7, const unsigned short* __restrict__ wt,
    float* __restrict__ out, int Btot)
{
    __shared__ unsigned short xbuf[8 * 16 * 136];   // per-wave 16x(<=128) act tiles
    __shared__ float          obuf[8 * 16 * 16];    // per-wave 16x16 f32 out tiles

    const int lane = threadIdx.x & 31;
    const int wave = threadIdx.x >> 5;
    unsigned short* xw = xbuf + wave * (16 * 136);
    float*          ob = obuf + wave * 256;
    const int pbase = blockIdx.x * 128 + wave * 16;
    if (pbase >= Btot) return;

    // ---- strain features: lanes 0..15 each build one particle row --------
    if (lane < 16) {
        const int p = pbase + lane;
        const float* Fp = F + (size_t)p * 9;
        float f0 = Fp[0], f1 = Fp[1], f2 = Fp[2];
        float f3 = Fp[3], f4 = Fp[4], f5 = Fp[5];
        float f6 = Fp[6], f7 = Fp[7], f8 = Fp[8];
        // FFt (symmetric)
        float a00 = f0*f0 + f1*f1 + f2*f2;
        float a01 = f0*f3 + f1*f4 + f2*f5;
        float a02 = f0*f6 + f1*f7 + f2*f8;
        float a11 = f3*f3 + f4*f4 + f5*f5;
        float a12 = f3*f6 + f4*f7 + f5*f8;
        float a22 = f6*f6 + f7*f7 + f8*f8;
        float det = f0*(f4*f8 - f5*f7) - f1*(f3*f8 - f5*f6) + f2*(f3*f7 - f4*f6);
        float J  = fmaxf(det, 1e-6f);
        float J1 = fmaxf(f0,  1e-6f);
        // singular values = sqrt(eigvals(FFt)); closed-form symmetric 3x3
        float q  = (a00 + a11 + a22) * (1.0f / 3.0f);
        float p1 = a01*a01 + a02*a02 + a12*a12;
        float b00 = a00 - q, b11 = a11 - q, b22 = a22 - q;
        float p2 = b00*b00 + b11*b11 + b22*b22 + 2.0f * p1;
        float pp = sqrtf(fmaxf(p2 * (1.0f / 6.0f), 1e-30f));
        float inv = 1.0f / pp;
        float detb = b00*(b11*b22 - a12*a12)
                   - a01*(a01*b22 - a12*a02)
                   + a02*(a01*a12 - b11*a02);
        float r = 0.5f * detb * inv * inv * inv;
        r = fminf(1.0f, fmaxf(-1.0f, r));
        float phi = acosf(r) * (1.0f / 3.0f);
        float e1 = q + 2.0f * pp * cosf(phi);
        float e3 = q + 2.0f * pp * cosf(phi + 2.0943951023931953f);
        float e2 = 3.0f * q - e1 - e3;
        float s1 = sqrtf(fmaxf(e1, 0.0f));
        float s2 = sqrtf(fmaxf(e2, 0.0f));
        float s3 = sqrtf(fmaxf(e3, 0.0f));

        unsigned short* xr = xw + lane * 136;
        xr[0] = f2bf(a00); xr[1] = f2bf(a01); xr[2] = f2bf(a02);
        xr[3] = f2bf(a01); xr[4] = f2bf(a11); xr[5] = f2bf(a12);
        xr[6] = f2bf(a02); xr[7] = f2bf(a12); xr[8] = f2bf(a22);
        xr[9]  = f2bf(logf(J));
        xr[10] = f2bf(s1); xr[11] = f2bf(s2); xr[12] = f2bf(s3);
        xr[13] = f2bf(J);
        xr[14] = f2bf(logf(J1)); xr[15] = f2bf(J1);
        const float* Cp = Cin + (size_t)p * 9;
#pragma unroll
        for (int e = 0; e < 9; ++e) xr[16 + e] = f2bf(Cp[e]);
        const float* lp = latent + (size_t)(*trajp) * 64;
#pragma unroll
        for (int e = 0; e < 64; ++e) xr[25 + e] = f2bf(lp[e]);
#pragma unroll
        for (int e = 89; e < 96; ++e) xr[e] = 0;
    }
    // Same-wave DS ordering guarantees all lanes see the rows below;
    // no workgroup barrier needed (tile is wave-private).

    const unsigned short* wt1 = wt;
    const unsigned short* wt2 = wt + 12288;
    const unsigned short* wt3 = wt + 12288 + 1 * 16384;
    const unsigned short* wt4 = wt + 12288 + 2 * 16384;
    const unsigned short* wt5 = wt + 12288 + 3 * 16384;
    const unsigned short* wt6 = wt + 12288 + 4 * 16384;
    const unsigned short* wt7 = wt + 94208;

    // Prefetch the next layer's weights while computing the current one
    // (emits gfx1250 global_prefetch_b8; weights are WGP$-hot after wave 0).
    __builtin_prefetch(wt2 + lane * 128, 0, 1);
    mlp_layer<3, true>(xw, wt1,  96, b1, lane);   // 89(pad 96) -> 128
    __builtin_prefetch(wt3 + lane * 128, 0, 1);
    mlp_layer<4, true>(xw, wt2, 128, b2, lane);
    __builtin_prefetch(wt4 + lane * 128, 0, 1);
    mlp_layer<4, true>(xw, wt3, 128, b3, lane);
    __builtin_prefetch(wt5 + lane * 128, 0, 1);
    mlp_layer<4, true>(xw, wt4, 128, b4, lane);
    __builtin_prefetch(wt6 + lane * 128, 0, 1);
    mlp_layer<4, true>(xw, wt5, 128, b5, lane);
    __builtin_prefetch(wt7 + lane * 64, 0, 1);
    mlp_layer<4, true>(xw, wt6, 128, b6, lane);

    // ---- layer 7: 128 -> 9 (padded N=16), then symmetrize ----------------
    {
        v16bf a[4];
#pragma unroll
        for (int kc = 0; kc < 4; ++kc) a[kc] = load_a_frag(xw, lane, kc);
        const int nl = lane & 15;
        const int mh = (lane >> 4) << 3;
        float bn = (nl < 9) ? b7[nl] : 0.0f;
        v8f c;
#pragma unroll
        for (int g = 0; g < 8; ++g) c[g] = bn;
#pragma unroll
        for (int kc = 0; kc < 4; ++kc) {
            v16bf b = load_b_frag(wt7, 128, 0, lane, kc);
            c = __builtin_amdgcn_wmma_f32_16x16x32_bf16(
                    false, a[kc], false, b, (short)0, c, false, false);
        }
#pragma unroll
        for (int g = 0; g < 8; ++g) ob[(g + mh) * 16 + nl] = c[g];

        // out = 0.5 * (O + O^T), 16 particles x 9 entries = 144 values
        for (int idx = lane; idx < 144; idx += 32) {
            int m = idx / 9, e = idx - m * 9;
            int i = e / 3,  j = e - i * 3;
            float v = 0.5f * (ob[m * 16 + e] + ob[m * 16 + j * 3 + i]);
            out[(size_t)(pbase + m) * 9 + e] = v;
        }
    }
}

// ---------------------------------------------------------------------------
extern "C" void kernel_launch(void* const* d_in, const int* in_sizes, int n_in,
                              void* d_out, int out_size, void* d_ws, size_t ws_size,
                              hipStream_t stream) {
    const float* F      = (const float*)d_in[0];
    const float* C      = (const float*)d_in[1];
    const int*   trajp  = (const int*)  d_in[2];
    const float* latent = (const float*)d_in[3];
    const float* W1 = (const float*)d_in[4];  const float* b1 = (const float*)d_in[5];
    const float* W2 = (const float*)d_in[6];  const float* b2 = (const float*)d_in[7];
    const float* W3 = (const float*)d_in[8];  const float* b3 = (const float*)d_in[9];
    const float* W4 = (const float*)d_in[10]; const float* b4 = (const float*)d_in[11];
    const float* W5 = (const float*)d_in[12]; const float* b5 = (const float*)d_in[13];
    const float* W6 = (const float*)d_in[14]; const float* b6 = (const float*)d_in[15];
    const float* W7 = (const float*)d_in[16]; const float* b7 = (const float*)d_in[17];

    unsigned short* wt = (unsigned short*)d_ws;   // needs ~193 KB
    float* out = (float*)d_out;

    const int Btot = in_sizes[0] / 9;             // 524288

    // 1) weights fp32 -> transposed/padded bf16 (96256 elements)
    wconv_kernel<<<(96256 + 255) / 256, 256, 0, stream>>>(W1, W2, W3, W4, W5, W6, W7, wt);

    // 2) fused feature + MLP + symmetrize; 128 particles per 256-thread block
    const int nblocks = (Btot + 127) / 128;
    stress_mlp_kernel<<<nblocks, 256, 0, stream>>>(
        F, C, trajp, latent, b1, b2, b3, b4, b5, b6, b7, wt, out, Btot);
}
